// CharLM_9560597201166
// MI455X (gfx1250) — compile-verified, matrix operand order checked by reading, not run on hardware
//
#include <hip/hip_runtime.h>
#include <hip/hip_bf16.h>

// ---------------- problem constants ----------------
#define E_DIM   1024
#define B_DIM   32
#define S_DIM   512
#define V_DIM   256
#define GATE_N  4096          // 4*E
#define NWG     64            // persistent workgroups; each owns E/NWG = 16 hidden units
#define ROWS    16384         // B*S

typedef __bf16 v16bf __attribute__((ext_vector_type(16)));
typedef __bf16 v8bf  __attribute__((ext_vector_type(8)));
typedef float  v8f   __attribute__((ext_vector_type(8)));

// ---------------- helpers ----------------
__device__ __forceinline__ unsigned short f2bf(float f) {
  unsigned u = __float_as_uint(f);
  unsigned r = u + 0x7FFFu + ((u >> 16) & 1u);   // RNE
  return (unsigned short)(r >> 16);
}
__device__ __forceinline__ float bf2f(unsigned short h) {
  return __uint_as_float(((unsigned)h) << 16);
}
__device__ __forceinline__ float sigmoidf_(float x) { return 1.f / (1.f + __expf(-x)); }

// A-fragment (16x32 bf16): lane<16 holds K{0..7,16..23}, lane>=16 holds K{8..15,24..31}
// p points at row base + ktile*32 + ((lane>>4)*8)
__device__ __forceinline__ v16bf load_a_frag(const unsigned short* p) {
  v8bf lo = *(const v8bf*)(p);
  v8bf hi = *(const v8bf*)(p + 16);
  v16bf a;
#pragma unroll
  for (int i = 0; i < 8; ++i) { a[i] = lo[i]; a[i + 8] = hi[i]; }
  return a;
}

__device__ __forceinline__ v8f wmma_bf16(v16bf a, v16bf b, v8f c) {
  return __builtin_amdgcn_wmma_f32_16x16x32_bf16(false, a, false, b, (short)0, c,
                                                 false, false);
}

// device-wide sense barrier over NWG workgroups (monotone counter)
__device__ __forceinline__ void grid_barrier(unsigned* cnt, unsigned* epoch) {
  __syncthreads();
  if (threadIdx.x == 0) {
    unsigned target = (++(*epoch)) * (unsigned)NWG;
    __threadfence();
    __hip_atomic_fetch_add(cnt, 1u, __ATOMIC_RELEASE, __HIP_MEMORY_SCOPE_AGENT);
    while (__hip_atomic_load(cnt, __ATOMIC_ACQUIRE, __HIP_MEMORY_SCOPE_AGENT) < target)
      __builtin_amdgcn_s_sleep(1);
  }
  __syncthreads();
}

// ---------------- setup kernels ----------------

// zero h ping-pong buffers and barrier counter
__global__ void clm_init(unsigned short* h0, unsigned short* h1, unsigned* cnt) {
  int i = blockIdx.x * 256 + threadIdx.x;
  if (i < 2 * B_DIM * E_DIM) { h0[i] = 0; h1[i] = 0; }
  if (i == 0) *cnt = 0u;
}

// F.normalize(embedding, dim=-1) -> bf16 copy
__global__ void clm_embnorm(const float* __restrict__ emb,
                            unsigned short* __restrict__ embbf) {
  __shared__ float red[256];
  int v = blockIdx.x, tid = threadIdx.x;
  float ss = 0.f;
  for (int e = tid; e < E_DIM; e += 256) {
    float x = emb[(size_t)v * E_DIM + e];
    ss += x * x;
  }
  red[tid] = ss; __syncthreads();
  for (int s = 128; s > 0; s >>= 1) {
    if (tid < s) red[tid] += red[tid + s];
    __syncthreads();
  }
  float denom = fmaxf(sqrtf(red[0]), 1e-12f);
  for (int e = tid; e < E_DIM; e += 256)
    embbf[(size_t)v * E_DIM + e] = f2bf(emb[(size_t)v * E_DIM + e] / denom);
}

// pack fp32 weight W[nrow][ldK], cols [colOff, colOff+1024) into WMMA B-frag tiles.
// grid = ntiles*32 blocks; tile element e = lane*16 + i.
__global__ void clm_pack_w(const float* __restrict__ W, int ldK, int colOff,
                           unsigned short* __restrict__ dst) {
  int ntile = blockIdx.x >> 5;
  int ktile = blockIdx.x & 31;
  for (int e = threadIdx.x; e < 512; e += 256) {
    int lane = e >> 4, i = e & 15;
    int n  = lane & 15;
    int kb = (lane >> 4) * 8;
    int k  = (i < 8) ? (kb + i) : (kb + 16 + (i - 8));
    int nrow = ntile * 16 + n;
    int kcol = colOff + ktile * 32 + k;
    dst[(size_t)blockIdx.x * 512 + e] = f2bf(W[(size_t)nrow * ldK + kcol]);
  }
}

// pack normalized embedding (bf16) as logits B-matrix: B[k][v] = emb_n[v][k]
__global__ void clm_pack_emb(const unsigned short* __restrict__ embbf,
                             unsigned short* __restrict__ dst) {
  int ntile = blockIdx.x >> 5;   // 16 n-tiles (V=256)
  int ktile = blockIdx.x & 31;   // 32 k-tiles (E=1024)
  for (int e = threadIdx.x; e < 512; e += 256) {
    int lane = e >> 4, i = e & 15;
    int n  = lane & 15;
    int kb = (lane >> 4) * 8;
    int k  = (i < 8) ? (kb + i) : (kb + 16 + (i - 8));
    int vrow = ntile * 16 + n;
    int kcol = ktile * 32 + k;
    dst[(size_t)blockIdx.x * 512 + e] = embbf[(size_t)vrow * E_DIM + kcol];
  }
}

// x[s][b][:] = emb_n[inputs[b][s]][:]  (bf16)
__global__ void clm_gather(const int* __restrict__ inputs,
                           const unsigned short* __restrict__ embbf,
                           unsigned short* __restrict__ x) {
  int bid = blockIdx.x;             // b*512 + s
  int b = bid >> 9, s = bid & 511;
  int tok = inputs[bid];
  const unsigned short* src = embbf + (size_t)tok * E_DIM;
  unsigned short* dstp = x + ((size_t)s * B_DIM + b) * E_DIM;
  for (int e = threadIdx.x; e < E_DIM; e += 256) dstp[e] = src[e];
}

__global__ void clm_bias(const float* bi0, const float* bh0,
                         const float* bi1, const float* bh1,
                         float* b0, float* b1) {
  int i = blockIdx.x * 256 + threadIdx.x;
  if (i < GATE_N) { b0[i] = bi0[i] + bh0[i]; b1[i] = bi1[i] + bh1[i]; }
}

// ---------------- persistent LSTM recurrence ----------------
// 64 WGs x 256 threads. WG wg owns hidden units [wg*16, wg*16+16).
// Wave w: mt = w>>2 (batch tile), g = w&3 (gate i/f/g/o) -> one 16x16 C tile.
__global__ __launch_bounds__(256) void clm_lstm(
    const unsigned short* __restrict__ x,     // [S][B][E] bf16
    const unsigned short* __restrict__ wx,    // packed [256][32][512]
    const unsigned short* __restrict__ wp,
    const unsigned short* __restrict__ whh0,
    const unsigned short* __restrict__ wih1,
    const unsigned short* __restrict__ whh1,
    const float* __restrict__ b0,
    const float* __restrict__ b1,
    unsigned short* __restrict__ h0buf,       // [2][B][E]
    unsigned short* __restrict__ h1buf,       // [2][B][E]
    unsigned short* __restrict__ outbf,       // [B*S][E]
    unsigned* __restrict__ barrier_cnt) {
  __shared__ float gates[4][B_DIM][16];
  __shared__ float c0s[B_DIM][16];
  __shared__ float c1s[B_DIM][16];

  const int tid  = threadIdx.x;
  const int wg   = blockIdx.x;
  const int wid  = tid >> 5;
  const int lane = tid & 31;
  const int mt   = wid >> 2;
  const int g    = wid & 3;
  const int ln   = lane & 15;
  const int kb   = (lane >> 4) * 8;
  const int mrow = mt * 16 + ln;          // batch row for A frags

  for (int p = tid; p < B_DIM * 16; p += 256) {
    c0s[p >> 4][p & 15] = 0.f;
    c1s[p >> 4][p & 15] = 0.f;
  }

  const float bias0 = b0[g * E_DIM + wg * 16 + ln];
  const float bias1 = b1[g * E_DIM + wg * 16 + ln];

  const size_t nt = (size_t)(g * 64 + wg);  // this wave's n-tile (of 256)
  const unsigned short* wxT   = wx   + nt * 32 * 512 + (size_t)lane * 16;
  const unsigned short* wpT   = wp   + nt * 32 * 512 + (size_t)lane * 16;
  const unsigned short* whh0T = whh0 + nt * 32 * 512 + (size_t)lane * 16;
  const unsigned short* wih1T = wih1 + nt * 32 * 512 + (size_t)lane * 16;
  const unsigned short* whh1T = whh1 + nt * 32 * 512 + (size_t)lane * 16;

  unsigned epoch = 0;

  for (int t = 0; t < S_DIM; ++t) {
    const int pp = t & 1, cc = pp ^ 1;    // ping-pong buffers
    const unsigned short* h0p = h0buf + (size_t)pp * B_DIM * E_DIM + (size_t)mrow * E_DIM;
    const unsigned short* h1p = h1buf + (size_t)pp * B_DIM * E_DIM + (size_t)mrow * E_DIM;
    const unsigned short* xa  = x + ((size_t)t * B_DIM + mrow) * E_DIM;

    // -------- layer 0: gates = [x_t, h1_prev] @ Wih0^T + h0_prev @ Whh0^T + b0
    v8f acc;
#pragma unroll
    for (int r = 0; r < 8; ++r) acc[r] = bias0;
#pragma unroll 8
    for (int kt = 0; kt < 32; ++kt)
      acc = wmma_bf16(load_a_frag(xa + kt * 32 + kb),
                      *(const v16bf*)(wxT + (size_t)kt * 512), acc);
#pragma unroll 8
    for (int kt = 0; kt < 32; ++kt)
      acc = wmma_bf16(load_a_frag(h1p + kt * 32 + kb),
                      *(const v16bf*)(wpT + (size_t)kt * 512), acc);
#pragma unroll 8
    for (int kt = 0; kt < 32; ++kt)
      acc = wmma_bf16(load_a_frag(h0p + kt * 32 + kb),
                      *(const v16bf*)(whh0T + (size_t)kt * 512), acc);
#pragma unroll
    for (int r = 0; r < 8; ++r)
      gates[g][mt * 16 + r + 8 * (lane >> 4)][ln] = acc[r];
    __syncthreads();

    for (int p = tid; p < B_DIM * 16; p += 256) {
      int b_ = p >> 4, u = p & 15;
      float iv = gates[0][b_][u], fv = gates[1][b_][u];
      float gv = gates[2][b_][u], ov = gates[3][b_][u];
      float cn = sigmoidf_(fv) * c0s[b_][u] + sigmoidf_(iv) * tanhf(gv);
      float hn = sigmoidf_(ov) * tanhf(cn);
      c0s[b_][u] = cn;
      h0buf[(size_t)cc * B_DIM * E_DIM + (size_t)b_ * E_DIM + wg * 16 + u] = f2bf(hn);
    }
    grid_barrier(barrier_cnt, &epoch);    // h0(new) visible everywhere

    // -------- layer 1: gates = h0_new @ Wih1^T + h1_prev @ Whh1^T + b1
    const unsigned short* h0c = h0buf + (size_t)cc * B_DIM * E_DIM + (size_t)mrow * E_DIM;
#pragma unroll
    for (int r = 0; r < 8; ++r) acc[r] = bias1;
#pragma unroll 8
    for (int kt = 0; kt < 32; ++kt)
      acc = wmma_bf16(load_a_frag(h0c + kt * 32 + kb),
                      *(const v16bf*)(wih1T + (size_t)kt * 512), acc);
#pragma unroll 8
    for (int kt = 0; kt < 32; ++kt)
      acc = wmma_bf16(load_a_frag(h1p + kt * 32 + kb),
                      *(const v16bf*)(whh1T + (size_t)kt * 512), acc);
#pragma unroll
    for (int r = 0; r < 8; ++r)
      gates[g][mt * 16 + r + 8 * (lane >> 4)][ln] = acc[r];
    __syncthreads();

    for (int p = tid; p < B_DIM * 16; p += 256) {
      int b_ = p >> 4, u = p & 15;
      float iv = gates[0][b_][u], fv = gates[1][b_][u];
      float gv = gates[2][b_][u], ov = gates[3][b_][u];
      float cn = sigmoidf_(fv) * c1s[b_][u] + sigmoidf_(iv) * tanhf(gv);
      float hn = sigmoidf_(ov) * tanhf(cn);
      c1s[b_][u] = cn;
      unsigned short hb = f2bf(hn);
      h1buf[(size_t)cc * B_DIM * E_DIM + (size_t)b_ * E_DIM + wg * 16 + u] = hb;
      outbf[((size_t)b_ * S_DIM + t) * E_DIM + wg * 16 + u] = hb;
    }
    grid_barrier(barrier_cnt, &epoch);    // h1(new) visible everywhere
  }
}

// ---------------- ScaleNorm (in place on bf16 outputs) ----------------
__global__ void clm_scalenorm(unsigned short* __restrict__ outbf,
                              const float* __restrict__ scale_p) {
  __shared__ float red[256];
  int r = blockIdx.x, tid = threadIdx.x;
  unsigned short* row = outbf + (size_t)r * E_DIM;
  float ss = 0.f;
  for (int e = tid; e < E_DIM; e += 256) { float v = bf2f(row[e]); ss += v * v; }
  red[tid] = ss; __syncthreads();
  for (int s = 128; s > 0; s >>= 1) {
    if (tid < s) red[tid] += red[tid + s];
    __syncthreads();
  }
  float sc = scale_p[0] / fmaxf(sqrtf(red[0]), 1e-5f);
  for (int e = tid; e < E_DIM; e += 256) row[e] = f2bf(bf2f(row[e]) * sc);
}

// ---------------- logits = outputs @ emb_n^T  (WMMA) ----------------
// block: 32 rows x 256 cols; 8 waves; wave w: mt = w&1, ntbase = (w>>1)*4
__global__ __launch_bounds__(256) void clm_logits(
    const unsigned short* __restrict__ outbf,  // [ROWS][E] bf16, scaled
    const unsigned short* __restrict__ embp,   // packed [16][32][512]
    float* __restrict__ logits) {              // [ROWS][V]
  const int tid = threadIdx.x, wid = tid >> 5, lane = tid & 31;
  const int mt = wid & 1, ntbase = (wid >> 1) * 4;
  const int ln = lane & 15, kb = (lane >> 4) * 8;
  const int rowbase = blockIdx.x * 32;
  const int mrow = rowbase + mt * 16 + ln;
  const unsigned short* arow = outbf + (size_t)mrow * E_DIM;

  v8f acc[4];
#pragma unroll
  for (int j = 0; j < 4; ++j)
#pragma unroll
    for (int r = 0; r < 8; ++r) acc[j][r] = 0.f;

#pragma unroll 4
  for (int kt = 0; kt < 32; ++kt) {
    v16bf a = load_a_frag(arow + kt * 32 + kb);
#pragma unroll
    for (int j = 0; j < 4; ++j) {
      const unsigned short* bp =
          embp + ((size_t)(ntbase + j) * 32 + kt) * 512 + (size_t)lane * 16;
      acc[j] = wmma_bf16(a, *(const v16bf*)bp, acc[j]);
    }
  }
#pragma unroll
  for (int j = 0; j < 4; ++j)
#pragma unroll
    for (int r = 0; r < 8; ++r) {
      int row = rowbase + mt * 16 + r + 8 * (lane >> 4);
      int col = (ntbase + j) * 16 + ln;
      logits[(size_t)row * V_DIM + col] = acc[j][r];
    }
}

// ---------------- loss ----------------
__global__ void clm_loss_row(const float* __restrict__ logits,
                             const int* __restrict__ targets,
                             float* __restrict__ rn, float* __restrict__ rs) {
  __shared__ float red[256];
  int r = blockIdx.x, tid = threadIdx.x;
  float x = logits[(size_t)r * V_DIM + tid];
  red[tid] = x; __syncthreads();
  for (int s = 128; s > 0; s >>= 1) {
    if (tid < s) red[tid] = fmaxf(red[tid], red[tid + s]);
    __syncthreads();
  }
  float mx = red[0]; __syncthreads();
  red[tid] = __expf(x - mx); __syncthreads();
  for (int s = 128; s > 0; s >>= 1) {
    if (tid < s) red[tid] += red[tid + s];
    __syncthreads();
  }
  float lse = mx + __logf(red[0]); __syncthreads();
  red[tid] = x; __syncthreads();
  for (int s = 128; s > 0; s >>= 1) {
    if (tid < s) red[tid] += red[tid + s];
    __syncthreads();
  }
  if (tid == 0) {
    rn[r] = lse - logits[(size_t)r * V_DIM + targets[r]];
    rs[r] = (float)V_DIM * lse - red[0];
  }
}

__global__ void clm_loss_final(const float* __restrict__ rn,
                               const float* __restrict__ rs,
                               float* __restrict__ out) {
  __shared__ double ra[256], rb[256];
  int tid = threadIdx.x;
  double a = 0.0, b = 0.0;
  for (int i = tid; i < ROWS; i += 256) { a += (double)rn[i]; b += (double)rs[i]; }
  ra[tid] = a; rb[tid] = b; __syncthreads();
  for (int s = 128; s > 0; s >>= 1) {
    if (tid < s) { ra[tid] += ra[tid + s]; rb[tid] += rb[tid + s]; }
    __syncthreads();
  }
  if (tid == 0) {
    double nll = ra[0], smooth = rb[0];
    double loss = 0.9 * nll + 0.1 * smooth / (double)V_DIM;
    out[0] = (float)(loss / (double)ROWS);
    out[1] = (float)loss;
    out[2] = (float)nll;
  }
}

// ---------------- launch ----------------
extern "C" void kernel_launch(void* const* d_in, const int* in_sizes, int n_in,
                              void* d_out, int out_size, void* d_ws, size_t ws_size,
                              hipStream_t stream) {
  (void)in_sizes; (void)n_in; (void)out_size; (void)ws_size;
  const int*   inputs  = (const int*)d_in[0];
  const int*   targets = (const int*)d_in[1];
  const float* emb     = (const float*)d_in[2];
  const float* scale   = (const float*)d_in[3];
  const float* w_ih0   = (const float*)d_in[4];
  const float* w_hh0   = (const float*)d_in[5];
  const float* b_ih0   = (const float*)d_in[6];
  const float* b_hh0   = (const float*)d_in[7];
  const float* w_ih1   = (const float*)d_in[8];
  const float* w_hh1   = (const float*)d_in[9];
  const float* b_ih1   = (const float*)d_in[10];
  const float* b_hh1   = (const float*)d_in[11];

  char* ws = (char*)d_ws;
  size_t off = 0;
  auto alloc = [&](size_t bytes) -> void* {
    void* p = (void*)(ws + off);
    off += (bytes + 255) & ~(size_t)255;
    return p;
  };
  float*          b0     = (float*)alloc(GATE_N * 4);
  float*          b1     = (float*)alloc(GATE_N * 4);
  unsigned*       cnt    = (unsigned*)alloc(256);
  unsigned short* h0buf  = (unsigned short*)alloc((size_t)2 * B_DIM * E_DIM * 2);
  unsigned short* h1buf  = (unsigned short*)alloc((size_t)2 * B_DIM * E_DIM * 2);
  unsigned short* embbf  = (unsigned short*)alloc((size_t)V_DIM * E_DIM * 2);
  unsigned short* embp   = (unsigned short*)alloc((size_t)16 * 32 * 512 * 2);
  unsigned short* wx     = (unsigned short*)alloc((size_t)256 * 32 * 512 * 2);
  unsigned short* wp     = (unsigned short*)alloc((size_t)256 * 32 * 512 * 2);
  unsigned short* whh0   = (unsigned short*)alloc((size_t)256 * 32 * 512 * 2);
  unsigned short* wih1   = (unsigned short*)alloc((size_t)256 * 32 * 512 * 2);
  unsigned short* whh1   = (unsigned short*)alloc((size_t)256 * 32 * 512 * 2);
  unsigned short* xbf    = (unsigned short*)alloc((size_t)S_DIM * B_DIM * E_DIM * 2);
  unsigned short* outbf  = (unsigned short*)alloc((size_t)ROWS * E_DIM * 2);
  float*          rn     = (float*)alloc((size_t)ROWS * 4);
  float*          rs     = (float*)alloc((size_t)ROWS * 4);

  float* outp   = (float*)d_out;
  float* logits = outp + 3;

  clm_init<<<256, 256, 0, stream>>>(h0buf, h1buf, cnt);
  clm_embnorm<<<V_DIM, 256, 0, stream>>>(emb, embbf);
  clm_pack_emb<<<16 * 32, 256, 0, stream>>>(embbf, embp);
  clm_gather<<<ROWS, 256, 0, stream>>>(inputs, embbf, xbf);
  clm_pack_w<<<256 * 32, 256, 0, stream>>>(w_ih0, 2 * E_DIM, 0, wx);
  clm_pack_w<<<256 * 32, 256, 0, stream>>>(w_ih0, 2 * E_DIM, E_DIM, wp);
  clm_pack_w<<<256 * 32, 256, 0, stream>>>(w_hh0, E_DIM, 0, whh0);
  clm_pack_w<<<256 * 32, 256, 0, stream>>>(w_ih1, E_DIM, 0, wih1);
  clm_pack_w<<<256 * 32, 256, 0, stream>>>(w_hh1, E_DIM, 0, whh1);
  clm_bias<<<16, 256, 0, stream>>>(b_ih0, b_hh0, b_ih1, b_hh1, b0, b1);

  clm_lstm<<<NWG, 256, 0, stream>>>(xbf, wx, wp, whh0, wih1, whh1, b0, b1,
                                    h0buf, h1buf, outbf, cnt);

  clm_scalenorm<<<ROWS, 256, 0, stream>>>(outbf, scale);
  clm_logits<<<ROWS / 32, 256, 0, stream>>>(outbf, embp, logits);
  clm_loss_row<<<ROWS, 256, 0, stream>>>(logits, targets, rn, rs);
  clm_loss_final<<<1, 256, 0, stream>>>(rn, rs, outp);
}
